// WindowAttention_44908178047410
// MI455X (gfx1250) — matrix operand order repaired and checked by reading, not compiled, not run on hardware
//
#include <hip/hip_runtime.h>
#include <hip/hip_bf16.h>

// ---------------------------------------------------------------------------
// Window attention (Swin), fused per-window kernel for gfx1250 (MI455X).
// All matmuls via v_wmma_f32_16x16x32_bf16 (wave32 WMMA).
// ---------------------------------------------------------------------------

typedef __attribute__((ext_vector_type(16))) __bf16 v16bf;
typedef __attribute__((ext_vector_type(8)))  float  v8f;

#define NTOK 64
#define CDIM 192
#define NH   6

// d_ws layout (bytes), all offsets 256-aligned
#define OFF_WQKVT  0u        // 576x192 bf16 = 221184 B (transposed, Q cols pre-scaled)
#define OFF_WPROJT 221184u   // 192x192 bf16 =  73728 B (transposed)
#define OFF_BQKV   294912u   // 576 f32 (Q part pre-scaled)
#define OFF_RPB    297216u   // 6x64x64 f32 relative-position bias

// LDS layout (bytes); strides in halves are multiples of 8 -> 16B aligned b128
#define QKV_STRIDE 392       // cols 0..383 used (Q: 0..191, K: 192..383)
#define LDS_XS   0u          // 64 x 200 halves = 25600 B (reused as Xo)
#define LDS_QKV  25600u      // 64 x 392 halves = 50176 B
#define LDS_VT   75776u      // 192 x 72 halves = 27648 B (V transposed: [h*32+d][token])
#define LDS_PS   103424u     // 8 waves x 16 x 72 halves = 18432 B
#define LDS_SIZE 121856u

union Frag {
  v16bf v;
  uint4 u[2];
};

union Pack8 {
  uint4  q;
  __bf16 h[8];
};

__device__ __forceinline__ v8f wmma_bf16(const Frag& a, const Frag& b, v8f c) {
  return __builtin_amdgcn_wmma_f32_16x16x32_bf16(false, a.v, false, b.v,
                                                 (short)0, c, false, false);
}

// ---------------------------------------------------------------------------
// Prep: transpose+convert weights to bf16, fold q-scale, expand rel-pos bias.
// ---------------------------------------------------------------------------
__global__ void prep_kernel(const float* __restrict__ qkv_w, const float* __restrict__ qkv_b,
                            const float* __restrict__ proj_w, const float* __restrict__ btab,
                            __bf16* __restrict__ WqkvT, __bf16* __restrict__ WprojT,
                            float* __restrict__ bqkv, float* __restrict__ rpb) {
  const float scale = 0.17677669529663687f;  // 1/sqrt(32)
  int idx = blockIdx.x * blockDim.x + threadIdx.x;
  if (idx < 576 * 192) {            // WqkvT[n][k] = qkv_w[k][n] (Q cols scaled)
    int n = idx / 192, k = idx - n * 192;
    float v = qkv_w[k * 576 + n];
    if (n < 192) v *= scale;
    WqkvT[idx] = (__bf16)v;
  }
  if (idx < 192 * 192) {            // WprojT[n][k] = proj_w[k][n]
    int n = idx / 192, k = idx - n * 192;
    WprojT[idx] = (__bf16)proj_w[k * 192 + n];
  }
  if (idx < 576) {
    float v = qkv_b[idx];
    if (idx < 192) v *= scale;
    bqkv[idx] = v;
  }
  if (idx < NH * 64 * 64) {         // rpb[h][q][k]
    int h = idx >> 12;
    int q = (idx >> 6) & 63;
    int k = idx & 63;
    int dy = (q >> 3) - (k >> 3);
    int dx = (q & 7) - (k & 7);
    int r  = (dy + 7) * 15 + (dx + 7);
    rpb[idx] = btab[r * NH + h];
  }
}

// ---------------------------------------------------------------------------
// Fused per-window attention. 1 block = 1 window, 256 threads = 8 wave32.
// ---------------------------------------------------------------------------
__global__ void __launch_bounds__(256)
win_attn_kernel(const float* __restrict__ X, const float* __restrict__ mask,
                const float* __restrict__ proj_b,
                const __bf16* __restrict__ WqkvT, const __bf16* __restrict__ WprojT,
                const float* __restrict__ bqkv, const float* __restrict__ rpb,
                float* __restrict__ out) {
  extern __shared__ char smem[];
  __bf16* Xs  = (__bf16*)(smem + LDS_XS);
  __bf16* QKV = (__bf16*)(smem + LDS_QKV);
  __bf16* Vt  = (__bf16*)(smem + LDS_VT);
  __bf16* Ps  = (__bf16*)(smem + LDS_PS);

  const int tid  = threadIdx.x;
  const int lane = tid & 31;
  const int wave = __builtin_amdgcn_readfirstlane(tid >> 5);   // scalar wave id
  const int l15  = lane & 15;
  const int lg   = lane >> 4;
  const int b    = blockIdx.x;
  const int wi   = b & 4095;                   // window index into mask (nW = 4096)

  const float* mp = mask + (size_t)wi * 4096;
  // prefetch this window's mask (16KB) while the QKV GEMM runs
  __builtin_prefetch(mp + tid * 16, 0, 3);

  // ---- stage A: load X window (64x192 f32) -> bf16 LDS ----
  const float4* gx = (const float4*)(X + (size_t)b * (NTOK * CDIM));
  #pragma unroll
  for (int j = 0; j < 12; ++j) {
    int i = tid + j * 256;                     // 3072 float4 total
    float4 v = gx[i];
    int f = i * 4;
    int row = f / 192;
    int col = f - row * 192;
    __bf16* d = Xs + row * 200 + col;
    d[0] = (__bf16)v.x; d[1] = (__bf16)v.y; d[2] = (__bf16)v.z; d[3] = (__bf16)v.w;
  }
  __syncthreads();

  // ---- stage B: QKV = Xs @ Wqkv + bqkv (64x576), Q pre-scaled ----
  // wave -> fixed row tile mt = wave/2; loop over 18 nt (parity wave&1).
  {
    const int mt = wave >> 1;                  // scalar
    const int np = wave & 1;                   // scalar
    const int m  = mt * 16 + l15;
    Frag a[6];                                 // A fragments loaded once (48 VGPRs)
    #pragma unroll
    for (int kc = 0; kc < 6; ++kc) {
      const __bf16* ap = Xs + m * 200 + kc * 32 + lg * 16;
      a[kc].u[0] = *(const uint4*)ap;
      a[kc].u[1] = *(const uint4*)(ap + 8);
    }
    for (int j = 0; j < 18; ++j) {
      const int nt = np + j * 2;               // scalar 0..35
      const int n  = nt * 16 + l15;
      v8f acc = {};
      #pragma unroll
      for (int kc = 0; kc < 6; ++kc) {
        Frag bb;
        const __bf16* bp = WqkvT + n * 192 + kc * 32 + lg * 16;
        bb.u[0] = *(const uint4*)bp;
        bb.u[1] = *(const uint4*)(bp + 8);
        acc = wmma_bf16(a[kc], bb, acc);
      }
      const float bias = bqkv[n];
      if (nt >= 24) {
        // V tile: store transposed only; 8 contiguous halves -> one b128 store
        Pack8 pk;
        #pragma unroll
        for (int r = 0; r < 8; ++r) pk.h[r] = (__bf16)(acc[r] + bias);
        *(uint4*)&Vt[(n - 384) * 72 + mt * 16 + lg * 8] = pk.q;
      } else {
        // Q/K tile: token-major scatter (read later as rows)
        #pragma unroll
        for (int r = 0; r < 8; ++r) {
          int row = mt * 16 + r + lg * 8;
          QKV[row * QKV_STRIDE + n] = (__bf16)(acc[r] + bias);
        }
      }
    }
  }
  __syncthreads();

  // ---- stage C: per (head, 16-row tile): S = Q K^T, softmax, O = P V ----
  for (int rb = wave; rb < 24; rb += 8) {
    const int h  = rb >> 2;                    // scalar
    const int mt = rb & 3;                     // scalar

    Frag aq;                                   // Q A-fragment, reused over nt
    const __bf16* qp = QKV + (mt * 16 + l15) * QKV_STRIDE + h * 32 + lg * 16;
    aq.u[0] = *(const uint4*)qp;
    aq.u[1] = *(const uint4*)(qp + 8);

    v8f s[4];
    #pragma unroll
    for (int nt = 0; nt < 4; ++nt) {
      Frag bk;                                 // K^T B-fragment: row=token, contiguous d
      const __bf16* kp = QKV + (nt * 16 + l15) * QKV_STRIDE + 192 + h * 32 + lg * 16;
      bk.u[0] = *(const uint4*)kp;
      bk.u[1] = *(const uint4*)(kp + 8);
      v8f z = {};
      s[nt] = wmma_bf16(aq, bk, z);
    }

    const float* rp = rpb + h * 4096;
    #pragma unroll
    for (int r = 0; r < 8; ++r) {
      int m = mt * 16 + r + lg * 8;
      #pragma unroll
      for (int nt = 0; nt < 4; ++nt) {
        int n = nt * 16 + l15;
        s[nt][r] += rp[m * 64 + n] + mp[m * 64 + n];
      }
    }

    // softmax over 64 cols; each row lives in one 16-lane half-group
    #pragma unroll
    for (int r = 0; r < 8; ++r) {
      float mx = fmaxf(fmaxf(s[0][r], s[1][r]), fmaxf(s[2][r], s[3][r]));
      #pragma unroll
      for (int off = 1; off < 16; off <<= 1)
        mx = fmaxf(mx, __shfl_xor(mx, off, 32));
      float sum = 0.f;
      #pragma unroll
      for (int nt = 0; nt < 4; ++nt) {
        float e = __expf(s[nt][r] - mx);
        s[nt][r] = e;
        sum += e;
      }
      #pragma unroll
      for (int off = 1; off < 16; off <<= 1)
        sum += __shfl_xor(sum, off, 32);
      float inv = __frcp_rn(sum);
      #pragma unroll
      for (int nt = 0; nt < 4; ++nt) s[nt][r] *= inv;
    }

    // C-layout -> A-layout via per-wave LDS scratch
    __bf16* psw = Ps + wave * (16 * 72);
    #pragma unroll
    for (int r = 0; r < 8; ++r) {
      int row = r + lg * 8;
      #pragma unroll
      for (int nt = 0; nt < 4; ++nt)
        psw[row * 72 + nt * 16 + l15] = (__bf16)s[nt][r];
    }

    // O = P @ V_h (16x32)
    #pragma unroll
    for (int ot = 0; ot < 2; ++ot) {
      v8f o = {};
      #pragma unroll
      for (int kc = 0; kc < 2; ++kc) {
        Frag ap, bv;
        const __bf16* pp = psw + l15 * 72 + kc * 32 + lg * 16;
        ap.u[0] = *(const uint4*)pp;
        ap.u[1] = *(const uint4*)(pp + 8);
        const __bf16* vp = Vt + (h * 32 + ot * 16 + l15) * 72 + kc * 32 + lg * 16;
        bv.u[0] = *(const uint4*)vp;
        bv.u[1] = *(const uint4*)(vp + 8);
        o = wmma_bf16(ap, bv, o);
      }
      #pragma unroll
      for (int r = 0; r < 8; ++r) {
        int row = mt * 16 + r + lg * 8;
        Xs[row * 200 + h * 32 + ot * 16 + l15] = (__bf16)o[r];   // Xo reuses Xs
      }
    }
  }
  __syncthreads();

  // ---- stage D: out = Xo @ proj_w + proj_b (64x192, f32 out) ----
  float* op = out + (size_t)b * (NTOK * CDIM);
  {
    const int mt = wave >> 1;                  // scalar
    const int np = wave & 1;                   // scalar
    const int m  = mt * 16 + l15;
    Frag a[6];                                 // A fragments loaded once
    #pragma unroll
    for (int kc = 0; kc < 6; ++kc) {
      const __bf16* ap = Xs + m * 200 + kc * 32 + lg * 16;
      a[kc].u[0] = *(const uint4*)ap;
      a[kc].u[1] = *(const uint4*)(ap + 8);
    }
    #pragma unroll
    for (int j = 0; j < 6; ++j) {
      const int nt = np + j * 2;               // scalar 0..11
      const int n  = nt * 16 + l15;
      v8f acc = {};
      #pragma unroll
      for (int kc = 0; kc < 6; ++kc) {
        Frag bb;
        const __bf16* bp = WprojT + n * 192 + kc * 32 + lg * 16;
        bb.u[0] = *(const uint4*)bp;
        bb.u[1] = *(const uint4*)(bp + 8);
        acc = wmma_bf16(a[kc], bb, acc);
      }
      const float bias = proj_b[n];
      #pragma unroll
      for (int r = 0; r < 8; ++r) {
        int row = mt * 16 + r + lg * 8;
        op[row * 192 + n] = acc[r] + bias;
      }
    }
  }
}

// ---------------------------------------------------------------------------
extern "C" void kernel_launch(void* const* d_in, const int* in_sizes, int n_in,
                              void* d_out, int out_size, void* d_ws, size_t ws_size,
                              hipStream_t stream) {
  const float* hidden = (const float*)d_in[0];
  const float* maskp  = (const float*)d_in[1];
  const float* qkv_w  = (const float*)d_in[2];
  const float* qkv_b  = (const float*)d_in[3];
  const float* proj_w = (const float*)d_in[4];
  const float* proj_b = (const float*)d_in[5];
  const float* btab   = (const float*)d_in[6];

  char* ws = (char*)d_ws;
  __bf16* WqkvT  = (__bf16*)(ws + OFF_WQKVT);
  __bf16* WprojT = (__bf16*)(ws + OFF_WPROJT);
  float*  bqkv   = (float*)(ws + OFF_BQKV);
  float*  rpb    = (float*)(ws + OFF_RPB);

  prep_kernel<<<432, 256, 0, stream>>>(qkv_w, qkv_b, proj_w, btab,
                                       WqkvT, WprojT, bqkv, rpb);

  const int B_ = in_sizes[0] / (NTOK * CDIM);  // 8192 windows
  const size_t lds = LDS_SIZE;                 // 119 KB; 2 blocks / 320KB WGP
  (void)hipFuncSetAttribute((const void*)win_attn_kernel,
                            hipFuncAttributeMaxDynamicSharedMemorySize, (int)lds);
  win_attn_kernel<<<B_, 256, lds, stream>>>(hidden, maskp, proj_b,
                                            WqkvT, WprojT, bqkv, rpb,
                                            (float*)d_out);
}
